// MambaBlock_39152921870378
// MI455X (gfx1250) — matrix operand order repaired and checked by reading, not compiled, not run on hardware
//
#include <hip/hip_runtime.h>
#include <hip/hip_bf16.h>

// ---------------------------------------------------------------------------
// Mamba block forward for MI455X (gfx1250, wave32, WMMA).
// GEMMs run as bf16 x bf16 -> f32 via V_WMMA_F32_16X16X32_BF16, with an
// explicitly software-pipelined K loop (prefetch next fragments under the
// current WMMAs).
// ---------------------------------------------------------------------------

#define B_SZ     2
#define L_SZ     1024
#define D_MODEL  768
#define D_INNER  1536
#define D_STATE  16
#define D_CONV   4
#define DT_RANK  48
#define M_TOK    (B_SZ * L_SZ)   // 2048 tokens

typedef __attribute__((ext_vector_type(16))) __bf16          v16bf;
typedef __attribute__((ext_vector_type(8)))  float           v8f;
typedef __attribute__((ext_vector_type(8)))  unsigned short  v8us;
typedef __attribute__((ext_vector_type(16))) unsigned short  v16us;

__device__ __forceinline__ unsigned short f32_to_bf16u(float f) {
    unsigned int u = __builtin_bit_cast(unsigned int, f);
    u += 0x7FFFu + ((u >> 16) & 1u);   // round-to-nearest-even
    return (unsigned short)(u >> 16);
}

// ---------------------------------------------------------------------------
// 1) RMSNorm over D_MODEL, emit bf16 normalized activations.
// ---------------------------------------------------------------------------
__global__ void rmsnorm_bf16_kernel(const float* __restrict__ x,
                                    const float* __restrict__ w,
                                    unsigned short* __restrict__ out) {
    __shared__ float red[256];
    const int m = blockIdx.x;
    const float* row = x + (size_t)m * D_MODEL;
    const int t = threadIdx.x;
    float v0 = row[t], v1 = row[t + 256], v2 = row[t + 512];
    red[t] = v0 * v0 + v1 * v1 + v2 * v2;
    __syncthreads();
    for (int off = 128; off > 0; off >>= 1) {
        if (t < off) red[t] += red[t + off];
        __syncthreads();
    }
    const float rn = rsqrtf(red[0] / (float)D_MODEL + 1e-5f);
    unsigned short* orow = out + (size_t)m * D_MODEL;
    orow[t]       = f32_to_bf16u(v0 * rn * w[t]);
    orow[t + 256] = f32_to_bf16u(v1 * rn * w[t + 256]);
    orow[t + 512] = f32_to_bf16u(v2 * rn * w[t + 512]);
}

// ---------------------------------------------------------------------------
// 2) Generic fp32 -> bf16 conversion (for weight matrices).
// ---------------------------------------------------------------------------
__global__ void cvt_bf16_kernel(const float* __restrict__ src,
                                unsigned short* __restrict__ dst, int n) {
    int i = blockIdx.x * 256 + threadIdx.x;
    if (i < n) dst[i] = f32_to_bf16u(src[i]);
}

// ---------------------------------------------------------------------------
// 3) bf16 WMMA GEMM:  C[M,N] = A[M,K] * W[N,K]^T  (+ optional addsrc[M,N])
//    8 waves / block; each wave owns a 16(M) x (16*JT)(N) tile with JT
//    V_WMMA_F32_16X16X32_BF16 accumulators, stepping K by 32.
//    Requirements: M % 16 == 0, N % (16*JT) == 0, K % 32 == 0.
//
//    Fragment packing uses CONTIGUOUS k runs (k = 16*kh + s) for A and B:
//      A lane(ml,kh): K in [kh*16, kh*16+8) and [kh*16+8, kh*16+16)  (2x16B)
//      B lane(ml,kh): K in [kh*16, kh*16+16)                        (1x32B)
//
//    The K loop is software-pipelined: fragments for step k+1 are requested
//    before the WMMAs of step k are issued, so loadcnt waits overlap math.
// ---------------------------------------------------------------------------
template <int JT, bool ADD>
__global__ void gemm_wmma_bf16(const unsigned short* __restrict__ A,  // [M,K]
                               const unsigned short* __restrict__ W,  // [N,K]
                               const float* __restrict__ addsrc,      // [M,N]
                               float* __restrict__ C,                 // [M,N]
                               int M, int N, int K) {
    const int lane = threadIdx.x & 31;
    const int wave = threadIdx.x >> 5;
    const int ntn  = N / (16 * JT);
    const int gw   = blockIdx.x * 8 + wave;
    const int tm   = (gw / ntn) * 16;
    const int tn   = (gw % ntn) * (16 * JT);
    if (tm >= M) return;

    const int ml = lane & 15;   // row-in-tile (A) / col-in-tile (B,C)
    const int kh = lane >> 4;   // K-half / M-half selector

    const unsigned short* arow  = A + (size_t)(tm + ml) * K + kh * 16;
    const unsigned short* wbase = W + (size_t)(tn + ml) * K + kh * 16;

    auto load_a = [&](int k0) -> v16bf {
        const v8us a0 = *(const v8us*)(arow + k0);
        const v8us a1 = *(const v8us*)(arow + k0 + 8);
        const v16us au = __builtin_shufflevector(a0, a1,
            0, 1, 2, 3, 4, 5, 6, 7, 8, 9, 10, 11, 12, 13, 14, 15);
        return __builtin_bit_cast(v16bf, au);
    };
    auto load_b = [&](int j, int k0) -> v16bf {
        const v16us bu = *(const v16us*)(wbase + (size_t)j * 16 * K + k0);
        return __builtin_bit_cast(v16bf, bu);
    };

    v8f acc[JT];
#pragma unroll
    for (int j = 0; j < JT; ++j)
#pragma unroll
        for (int v = 0; v < 8; ++v) acc[j][v] = 0.0f;

    // Prologue: fragments for k = 0.
    v16bf afrag = load_a(0);
    v16bf bfrag[JT];
#pragma unroll
    for (int j = 0; j < JT; ++j) bfrag[j] = load_b(j, 0);

    // Pipelined main loop: issue next loads, then consume current fragments.
#pragma unroll 2
    for (int k0 = 0; k0 + 32 < K; k0 += 32) {
        const v16bf anext = load_a(k0 + 32);
        v16bf bnext[JT];
#pragma unroll
        for (int j = 0; j < JT; ++j) bnext[j] = load_b(j, k0 + 32);
#pragma unroll
        for (int j = 0; j < JT; ++j)
            acc[j] = __builtin_amdgcn_wmma_f32_16x16x32_bf16(
                false, afrag, false, bfrag[j], (short)0, acc[j], false, false);
        afrag = anext;
#pragma unroll
        for (int j = 0; j < JT; ++j) bfrag[j] = bnext[j];
    }
    // Epilogue: last K step.
#pragma unroll
    for (int j = 0; j < JT; ++j)
        acc[j] = __builtin_amdgcn_wmma_f32_16x16x32_bf16(
            false, afrag, false, bfrag[j], (short)0, acc[j], false, false);

    // C/D layout: VGPR v -> M = tm + v + kh*8; N = tn + j*16 + ml.
#pragma unroll
    for (int j = 0; j < JT; ++j) {
        const int col = tn + j * 16 + ml;
#pragma unroll
        for (int v = 0; v < 8; ++v) {
            const int row = tm + v + kh * 8;
            const size_t idx = (size_t)row * N + col;
            float val = acc[j][v];
            if (ADD) val += addsrc[idx];
            C[idx] = val;
        }
    }
}

// ---------------------------------------------------------------------------
// 4) Depthwise causal conv (k=4) over time + bias + SiLU -> u (f32 + bf16).
// ---------------------------------------------------------------------------
__global__ void conv_silu_kernel(const float* __restrict__ xz,
                                 const float* __restrict__ cw,   // [D_INNER,1,4]
                                 const float* __restrict__ cb,   // [D_INNER]
                                 float* __restrict__ u,
                                 unsigned short* __restrict__ ubf) {
    const int idx = blockIdx.x * 256 + threadIdx.x;
    if (idx >= M_TOK * D_INNER) return;
    const int d = idx % D_INNER;
    const int m = idx / D_INNER;
    const int l = m % L_SZ;
    const int b = m / L_SZ;
    float acc = cb[d];
#pragma unroll
    for (int j = 0; j < D_CONV; ++j) {
        const int ls = l + j - (D_CONV - 1);
        if (ls >= 0)
            acc = __builtin_fmaf(xz[(size_t)(b * L_SZ + ls) * (2 * D_INNER) + d],
                                 cw[d * D_CONV + j], acc);
    }
    const float s = acc / (1.0f + __expf(-acc));   // SiLU
    u[idx]   = s;
    ubf[idx] = f32_to_bf16u(s);
}

// ---------------------------------------------------------------------------
// 5) delta = softplus(dt @ dt_proj_w^T + dt_proj_b).  K=48 is too thin for
//    WMMA; stage the 48-wide dt row in LDS and do fp32 FMAs.
// ---------------------------------------------------------------------------
__global__ void delta_kernel(const float* __restrict__ xdbl,   // [M,80]
                             const float* __restrict__ dtw,    // [D_INNER,48]
                             const float* __restrict__ dtb,    // [D_INNER]
                             float* __restrict__ delta) {
    __shared__ float dtl[DT_RANK];
    const int m  = blockIdx.x / (D_INNER / 256);
    const int d0 = (blockIdx.x % (D_INNER / 256)) * 256;
    if (threadIdx.x < DT_RANK) dtl[threadIdx.x] = xdbl[(size_t)m * 80 + threadIdx.x];
    __syncthreads();
    const int d = d0 + threadIdx.x;
    float acc = dtb[d];
#pragma unroll
    for (int r = 0; r < DT_RANK; ++r)
        acc = __builtin_fmaf(dtl[r], dtw[d * DT_RANK + r], acc);
    const float sp = (acc > 20.0f) ? acc : logf(1.0f + __expf(acc));
    delta[(size_t)m * D_INNER + d] = sp;
}

// ---------------------------------------------------------------------------
// 6) Selective scan + SiLU(z) gating. One thread per (b,d); 16 states in
//    registers; sequential over L (true recurrence). Emits bf16 y.
// ---------------------------------------------------------------------------
__global__ void scan_kernel(const float* __restrict__ delta,  // [M,D_INNER]
                            const float* __restrict__ u,      // [M,D_INNER]
                            const float* __restrict__ xdbl,   // [M,80]
                            const float* __restrict__ xz,     // [M,2*D_INNER]
                            const float* __restrict__ A_log,  // [D_INNER,16]
                            const float* __restrict__ Dp,     // [D_INNER]
                            unsigned short* __restrict__ ybf) {
    const int idx = blockIdx.x * 256 + threadIdx.x;
    const int d = idx % D_INNER;
    const int b = idx / D_INNER;
    if (b >= B_SZ) return;

    float An[D_STATE], h[D_STATE];
#pragma unroll
    for (int n = 0; n < D_STATE; ++n) {
        An[n] = -__expf(A_log[d * D_STATE + n]);
        h[n]  = 0.0f;
    }
    const float dpar = Dp[d];

    for (int l = 0; l < L_SZ; ++l) {
        const int m = b * L_SZ + l;
        const float dv = delta[(size_t)m * D_INNER + d];
        const float uv = u[(size_t)m * D_INNER + d];
        const float* row = xdbl + (size_t)m * 80;
        const float dbu = dv * uv;
        float y = 0.0f;
#pragma unroll
        for (int n = 0; n < D_STATE; ++n) {
            const float dA = __expf(dv * An[n]);               // v_exp_f32
            h[n] = __builtin_fmaf(dA, h[n], dbu * row[DT_RANK + n]);
            y    = __builtin_fmaf(h[n], row[DT_RANK + D_STATE + n], y);
        }
        y = __builtin_fmaf(uv, dpar, y);
        const float z  = xz[(size_t)m * (2 * D_INNER) + D_INNER + d];
        const float sz = z / (1.0f + __expf(-z));              // SiLU gate
        ybf[(size_t)m * D_INNER + d] = f32_to_bf16u(y * sz);
    }
}

// ---------------------------------------------------------------------------
// Host-side launch
// ---------------------------------------------------------------------------
extern "C" void kernel_launch(void* const* d_in, const int* in_sizes, int n_in,
                              void* d_out, int out_size, void* d_ws, size_t ws_size,
                              hipStream_t stream) {
    (void)in_sizes; (void)n_in; (void)out_size; (void)ws_size;

    const float* hidden     = (const float*)d_in[0];   // [B,L,768]
    const float* norm_w     = (const float*)d_in[1];   // [768]
    const float* in_proj_w  = (const float*)d_in[2];   // [3072,768]
    const float* conv_w     = (const float*)d_in[3];   // [1536,1,4]
    const float* conv_b     = (const float*)d_in[4];   // [1536]
    const float* x_proj_w   = (const float*)d_in[5];   // [80,1536]
    const float* dt_proj_w  = (const float*)d_in[6];   // [1536,48]
    const float* dt_proj_b  = (const float*)d_in[7];   // [1536]
    const float* A_log      = (const float*)d_in[8];   // [1536,16]
    const float* Dparam     = (const float*)d_in[9];   // [1536]
    const float* out_proj_w = (const float*)d_in[10];  // [768,1536]
    float* out = (float*)d_out;                        // [B,L,768] fp32

    // Workspace carve-out (256B aligned)
    char* ws = (char*)d_ws;
    size_t off = 0;
    auto carve = [&](size_t bytes) -> void* {
        void* p = ws + off;
        off = (off + bytes + 255) & ~(size_t)255;
        return p;
    };
    unsigned short* xn_bf   = (unsigned short*)carve((size_t)M_TOK * D_MODEL * 2);
    unsigned short* win_bf  = (unsigned short*)carve((size_t)2 * D_INNER * D_MODEL * 2);
    unsigned short* wx_bf   = (unsigned short*)carve((size_t)(DT_RANK + 2 * D_STATE) * D_INNER * 2);
    unsigned short* wout_bf = (unsigned short*)carve((size_t)D_MODEL * D_INNER * 2);
    float*          xz      = (float*)carve((size_t)M_TOK * 2 * D_INNER * 4);
    float*          u_f     = (float*)carve((size_t)M_TOK * D_INNER * 4);
    unsigned short* u_bf    = (unsigned short*)carve((size_t)M_TOK * D_INNER * 2);
    float*          xdbl    = (float*)carve((size_t)M_TOK * 80 * 4);
    float*          delta   = (float*)carve((size_t)M_TOK * D_INNER * 4);
    unsigned short* y_bf    = (unsigned short*)carve((size_t)M_TOK * D_INNER * 2);

    // 1) RMSNorm -> bf16 activations
    rmsnorm_bf16_kernel<<<M_TOK, 256, 0, stream>>>(hidden, norm_w, xn_bf);

    // 2) Weight conversions to bf16
    {
        int n = 2 * D_INNER * D_MODEL;          // in_proj_w
        cvt_bf16_kernel<<<(n + 255) / 256, 256, 0, stream>>>(in_proj_w, win_bf, n);
        n = (DT_RANK + 2 * D_STATE) * D_INNER;  // x_proj_w
        cvt_bf16_kernel<<<(n + 255) / 256, 256, 0, stream>>>(x_proj_w, wx_bf, n);
        n = D_MODEL * D_INNER;                  // out_proj_w
        cvt_bf16_kernel<<<(n + 255) / 256, 256, 0, stream>>>(out_proj_w, wout_bf, n);
    }

    auto gemm_blocks = [](int M, int N, int jt) {
        const int tiles = (M / 16) * (N / (16 * jt));
        return (tiles + 7) / 8;
    };

    // 3) in_proj: xz[2048,3072] = xn[2048,768] * in_proj_w^T   (JT=4)
    gemm_wmma_bf16<4, false><<<gemm_blocks(M_TOK, 2 * D_INNER, 4), 256, 0, stream>>>(
        xn_bf, win_bf, nullptr, xz, M_TOK, 2 * D_INNER, D_MODEL);

    // 4) depthwise causal conv + SiLU -> u
    {
        const int n = M_TOK * D_INNER;
        conv_silu_kernel<<<(n + 255) / 256, 256, 0, stream>>>(xz, conv_w, conv_b, u_f, u_bf);
    }

    // 5) x_proj: xdbl[2048,80] = u * x_proj_w^T   (N=80 -> JT=1, 5 tiles)
    gemm_wmma_bf16<1, false><<<gemm_blocks(M_TOK, DT_RANK + 2 * D_STATE, 1), 256, 0, stream>>>(
        u_bf, wx_bf, nullptr, xdbl, M_TOK, DT_RANK + 2 * D_STATE, D_INNER);

    // 6) delta = softplus(dt @ dt_proj_w^T + b)
    delta_kernel<<<M_TOK * (D_INNER / 256), 256, 0, stream>>>(xdbl, dt_proj_w, dt_proj_b, delta);

    // 7) selective scan + SiLU(z) gating -> y (bf16)
    scan_kernel<<<(B_SZ * D_INNER) / 256, 256, 0, stream>>>(
        delta, u_f, xdbl, xz, A_log, Dparam, y_bf);

    // 8) out_proj with fused residual: out = y * out_proj_w^T + hidden  (JT=4)
    gemm_wmma_bf16<4, true><<<gemm_blocks(M_TOK, D_MODEL, 4), 256, 0, stream>>>(
        y_bf, wout_bf, hidden, out, M_TOK, D_MODEL, D_INNER);
}
